// GraphConv_77232101916990
// MI455X (gfx1250) — compile-verified, matrix operand order checked by reading, not run on hardware
//
#include <hip/hip_runtime.h>

#define NN 8192
#define DD 64
#define OUTSTRIDE 512
#define KTILE 128
#define LDS_STRIDE 136   // 128 + 8 pad (bf16 elems): 272 B rows -> conflict-free

typedef __attribute__((ext_vector_type(4)))  float        f32x4;
typedef __attribute__((ext_vector_type(4)))  unsigned int u32x4;
typedef __attribute__((ext_vector_type(8)))  float        v8f;
typedef __attribute__((ext_vector_type(8)))  __bf16       v8bf;
typedef __attribute__((ext_vector_type(16))) __bf16       v16bf;

// GCC-style <4 x i32> to match the async-to-LDS builtin's parameter type
typedef int v4i_gcc __attribute__((vector_size(16)));
typedef __attribute__((address_space(1))) v4i_gcc glob_v4i;
typedef __attribute__((address_space(3))) v4i_gcc locl_v4i;

__device__ __forceinline__ unsigned short f2bf(float f) {
  unsigned u = __float_as_uint(f);
  u += 0x7FFFu + ((u >> 16) & 1u);          // round-to-nearest-even
  return (unsigned short)(u >> 16);
}

// async copy of 16 bytes global -> LDS (ASYNCcnt-tracked)
__device__ __forceinline__ void async_b128(const unsigned short* g, unsigned short* l) {
#if __has_builtin(__builtin_amdgcn_global_load_async_to_lds_b128)
  __builtin_amdgcn_global_load_async_to_lds_b128(
      (glob_v4i*)g, (locl_v4i*)l, 0, 0);
#else
  unsigned lds_off = (unsigned)(unsigned long long)
      (__attribute__((address_space(3))) unsigned short*)l;
  asm volatile("global_load_async_to_lds_b128 %0, %1, off"
               :: "v"(lds_off), "v"(g) : "memory");
#endif
}

// Barrier for the producer/consumer LDS handoff:
// waits only DScnt (our ds_stores) and ASYNCcnt (our async B copies) --
// deliberately does NOT drain LOADcnt, so A-prefetch global loads stay in
// flight across the stage boundary (unlike __syncthreads()).
__device__ __forceinline__ void stage_sync() {
  asm volatile(
      "s_wait_dscnt 0x0\n\t"
      "s_wait_asynccnt 0x0\n\t"
      "s_barrier_signal -1\n\t"
      "s_barrier_wait -1"
      ::: "memory");
}

// convert 16 fp32 -> 16 bf16 and store as 2 x b128 to LDS
__device__ __forceinline__ void cvt_store16(unsigned short* dst,
                                            f32x4 a, f32x4 b, f32x4 c, f32x4 d) {
  unsigned short h[16];
  h[0]=f2bf(a[0]); h[1]=f2bf(a[1]); h[2]=f2bf(a[2]); h[3]=f2bf(a[3]);
  h[4]=f2bf(b[0]); h[5]=f2bf(b[1]); h[6]=f2bf(b[2]); h[7]=f2bf(b[3]);
  h[8]=f2bf(c[0]); h[9]=f2bf(c[1]); h[10]=f2bf(c[2]); h[11]=f2bf(c[3]);
  h[12]=f2bf(d[0]); h[13]=f2bf(d[1]); h[14]=f2bf(d[2]); h[15]=f2bf(d[3]);
  u32x4 p0, p1;
#pragma unroll
  for (int j = 0; j < 4; ++j) {
    p0[j] = (unsigned)h[2*j]     | ((unsigned)h[2*j+1] << 16);
    p1[j] = (unsigned)h[8+2*j]   | ((unsigned)h[9+2*j] << 16);
  }
  *(u32x4*)(dst)     = p0;
  *(u32x4*)(dst + 8) = p1;
}

struct GemmBatch {
  const float* A[4];
  const unsigned short* XT[4];
  float* Y[4];
  const float* W[4];
};

// ---------------------------------------------------------------------------
// 4 fused GEMMs: Y_i[N,64] = A_i[N,N] @ X_i[N,64]  (X_i pre-transposed bf16)
// 1024 blocks: blockIdx>>8 selects GEMM, &255 selects 32-row M block.
// Double-buffered LDS, async-to-LDS B operand, single barrier per K stage.
// ---------------------------------------------------------------------------
__global__ __launch_bounds__(256) void gemm4_wmma_bf16(GemmBatch gb)
{
  const int which = blockIdx.x >> 8;
  const int blk   = blockIdx.x & 255;
  const float*          A  = gb.A[which];
  const unsigned short* XT = gb.XT[which];
  float*                Y  = gb.Y[which];
  const float*          W  = gb.W[which];

  __shared__ unsigned short Albuf[2][32 * LDS_STRIDE];
  __shared__ unsigned short Blbuf[2][64 * LDS_STRIDE];

  const int t    = threadIdx.x;
  const int lane = t & 31;
  const int wave = t >> 5;
  const int wm   = wave >> 2;                  // 0..1
  const int wn   = wave & 3;                   // 0..3
  const int row0 = blk * 32;

  // cooperative-load coordinates
  const int ar = t >> 3;                       // 0..31 : A row, 16 floats each
  const int ak = (t & 7) * 16;
  const int br = t >> 2;                       // 0..63 : B row (n), 32 bf16 each
  const int bk = (t & 3) * 32;

  const float*          aptr = A  + (size_t)(row0 + ar) * NN + ak;
  const unsigned short* bptr = XT + (size_t)br * NN + bk;

  // WMMA operand-read coordinates
  const int arow  = wm * 16 + (lane & 15);
  const int bnrow = wn * 16 + (lane & 15);
  const int khalf = (lane >> 4) * 8;           // 0 or 8

  v8f acc = {0.f, 0.f, 0.f, 0.f, 0.f, 0.f, 0.f, 0.f};
  const int NSTAGE = NN / KTILE;               // 64

  // ---- prologue: stage 0 into buffer 0, prefetch A(1) into regs ----
#pragma unroll
  for (int c = 0; c < 4; ++c)
    async_b128(bptr + c * 8, &Blbuf[0][br * LDS_STRIDE + bk + c * 8]);

  f32x4 ra = *(const f32x4*)(aptr + 0);
  f32x4 rb = *(const f32x4*)(aptr + 4);
  f32x4 rc = *(const f32x4*)(aptr + 8);
  f32x4 rd = *(const f32x4*)(aptr + 12);
  cvt_store16(&Albuf[0][ar * LDS_STRIDE + ak], ra, rb, rc, rd);

  ra = *(const f32x4*)(aptr + KTILE + 0);
  rb = *(const f32x4*)(aptr + KTILE + 4);
  rc = *(const f32x4*)(aptr + KTILE + 8);
  rd = *(const f32x4*)(aptr + KTILE + 12);

  stage_sync();

  for (int s = 0; s < NSTAGE; ++s) {
    const int cur = s & 1;
    const int oth = cur ^ 1;

    // producers for stage s+1 (targets the buffer NOT consumed this stage)
    if (s + 1 < NSTAGE) {
      const unsigned short* bg = bptr + (size_t)(s + 1) * KTILE;
#pragma unroll
      for (int c = 0; c < 4; ++c)
        async_b128(bg + c * 8, &Blbuf[oth][br * LDS_STRIDE + bk + c * 8]);
      cvt_store16(&Albuf[oth][ar * LDS_STRIDE + ak], ra, rb, rc, rd);
      if (s + 2 < NSTAGE) {
        const float* ag = aptr + (size_t)(s + 2) * KTILE;
        ra = *(const f32x4*)(ag + 0);
        rb = *(const f32x4*)(ag + 4);
        rc = *(const f32x4*)(ag + 8);
        rd = *(const f32x4*)(ag + 12);
      }
    }

    // consume stage s: 4 x WMMA over K=128
    const unsigned short* Ab = &Albuf[cur][arow * LDS_STRIDE];
    const unsigned short* Bb = &Blbuf[cur][bnrow * LDS_STRIDE];
#pragma unroll
    for (int ks = 0; ks < 4; ++ks) {
      const int kb = ks * 32 + khalf;
      v8bf a0 = *(const v8bf*)(Ab + kb);
      v8bf a1 = *(const v8bf*)(Ab + kb + 16);
      v8bf b0 = *(const v8bf*)(Bb + kb);
      v8bf b1 = *(const v8bf*)(Bb + kb + 16);
      v16bf av = __builtin_shufflevector(a0, a1, 0,1,2,3,4,5,6,7,8,9,10,11,12,13,14,15);
      v16bf bv = __builtin_shufflevector(b0, b1, 0,1,2,3,4,5,6,7,8,9,10,11,12,13,14,15);
      acc = __builtin_amdgcn_wmma_f32_16x16x32_bf16(
          false, av, false, bv, (short)0, acc, false, false);
    }

    if (s + 1 < NSTAGE) {
      stage_sync();
    }
  }

  // C/D layout: vgpr v: lanes 0-15 -> M=v, lanes 16-31 -> M=8+v ; N = lane%16
  const int m0 = row0 + wm * 16 + (lane >> 4) * 8;
  const int n  = wn * 16 + (lane & 15);
  float* yp = Y + (size_t)m0 * DD + n;
  if (W) {
    const float* wp = W + (size_t)m0 * DD + n;
#pragma unroll
    for (int v = 0; v < 8; ++v) yp[(size_t)v * DD] = acc[v] * wp[(size_t)v * DD];
  } else {
#pragma unroll
    for (int v = 0; v < 8; ++v) yp[(size_t)v * DD] = acc[v];
  }
}

// ---------------------------------------------------------------------------
// 4 fused transposes: XT_i[c][k] = bf16(X_i[k][c]); XT_i at XT0 + i*N*64
// ---------------------------------------------------------------------------
struct Ptr4 { const float* p[4]; };

__global__ __launch_bounds__(256) void transpose4_to_bf16(
    Ptr4 xs, unsigned short* __restrict__ XT0)
{
  const int which = blockIdx.x >> 7;
  const int b     = blockIdx.x & 127;
  const float* X = xs.p[which];
  unsigned short* XT = XT0 + (size_t)which * NN * DD;

  const int i  = b * 256 + threadIdx.x;
  const int kc = i >> 6;
  const int c  = i & 63;
  const int k0 = kc * 16;
  unsigned short h[16];
#pragma unroll
  for (int j = 0; j < 16; ++j) h[j] = f2bf(X[(size_t)(k0 + j) * DD + c]);
  u32x4 lo, hi;
#pragma unroll
  for (int j = 0; j < 4; ++j) {
    lo[j] = (unsigned)h[2 * j]     | ((unsigned)h[2 * j + 1] << 16);
    hi[j] = (unsigned)h[8 + 2 * j] | ((unsigned)h[9 + 2 * j] << 16);
  }
  u32x4* p = (u32x4*)(XT + (size_t)c * NN + k0);
  p[0] = lo;
  p[1] = hi;
}

// ---------------------------------------------------------------------------
// dst[row, 0:64] = row / max(||row||2, 1e-12), dst row stride OUTSTRIDE
// ---------------------------------------------------------------------------
__global__ __launch_bounds__(256) void l2norm_write(
    const float* __restrict__ src, float* __restrict__ dst)
{
  const int lane = threadIdx.x & 31;
  const int row  = blockIdx.x * 8 + (threadIdx.x >> 5);
  const float* s = src + (size_t)row * DD;
  float x0 = s[lane], x1 = s[lane + 32];
  float ss = x0 * x0 + x1 * x1;
#pragma unroll
  for (int off = 16; off > 0; off >>= 1) ss += __shfl_xor(ss, off, 32);
  const float inv = 1.0f / fmaxf(sqrtf(ss), 1e-12f);
  float* d = dst + (size_t)row * OUTSTRIDE;
  d[lane]      = x0 * inv;
  d[lane + 32] = x1 * inv;
}

// ---------------------------------------------------------------------------
// wp1[n][d] = 1 + sum_f latent[f][d] * w[n][f]
// ---------------------------------------------------------------------------
__global__ __launch_bounds__(256) void make_wp1(
    const float* __restrict__ latent, const float* __restrict__ w,
    float* __restrict__ out)
{
  const int i = blockIdx.x * 256 + threadIdx.x;
  const int n = i >> 6, d = i & 63;
  float acc = 1.0f;
#pragma unroll
  for (int f = 0; f < 4; ++f) acc += latent[f * 64 + d] * w[n * 4 + f];
  out[i] = acc;
}

// ---------------------------------------------------------------------------
// sum of distance correlations over the 6 latent-factor pairs (tiny, 1 block)
// ---------------------------------------------------------------------------
__global__ void cor_kernel(const float* __restrict__ latent,
                           float* __restrict__ cor_out)
{
  __shared__ float t1[64], t2[64], ca[64], cb[64], r0[64], r1[64], r2[64];
  const int q = threadIdx.x;
  float acc = 0.0f;
  for (int i = 0; i < 4; ++i) {
    for (int j = i + 1; j < 4; ++j) {
      __syncthreads();
      t1[q] = latent[i * 64 + q];
      t2[q] = latent[j * 64 + q];
      __syncthreads();
      const float t1q = t1[q], t2q = t2[q];
      float sa = 0.f, sb = 0.f;
      for (int p = 0; p < 64; ++p) {
        float da = t1[p] - t1q, db = t2[p] - t2q;
        sa += sqrtf(da * da + 1e-8f);
        sb += sqrtf(db * db + 1e-8f);
      }
      ca[q] = sa; cb[q] = sb;
      __syncthreads();
      float ta = 0.f, tb = 0.f;
      for (int p = 0; p < 64; ++p) { ta += ca[p]; tb += cb[p]; }
      const float ma = ta / 4096.0f, mb = tb / 4096.0f;
      const float cmaq = sa / 64.0f, cmbq = sb / 64.0f;
      float sab = 0.f, saa = 0.f, sbb = 0.f;
      for (int p = 0; p < 64; ++p) {
        float da = t1[p] - t1q, db = t2[p] - t2q;
        float Av = sqrtf(da * da + 1e-8f) - ca[p] / 64.0f - cmaq + ma;
        float Bv = sqrtf(db * db + 1e-8f) - cb[p] / 64.0f - cmbq + mb;
        sab += Av * Bv; saa += Av * Av; sbb += Bv * Bv;
      }
      r0[q] = sab; r1[q] = saa; r2[q] = sbb;
      __syncthreads();
      if (q == 0) {
        float Sab = 0.f, Saa = 0.f, Sbb = 0.f;
        for (int p = 0; p < 64; ++p) { Sab += r0[p]; Saa += r1[p]; Sbb += r2[p]; }
        float dab = sqrtf(fmaxf(Sab / 4096.0f, 0.f) + 1e-8f);
        float daa = sqrtf(fmaxf(Saa / 4096.0f, 0.f) + 1e-8f);
        float dbb = sqrtf(fmaxf(Sbb / 4096.0f, 0.f) + 1e-8f);
        acc += dab / sqrtf(daa * dbb + 1e-8f);
      }
    }
  }
  if (q == 0) *cor_out = acc;
}

// ---------------------------------------------------------------------------
extern "C" void kernel_launch(void* const* d_in, const int* in_sizes, int n_in,
                              void* d_out, int out_size, void* d_ws, size_t ws_size,
                              hipStream_t stream)
{
  (void)in_sizes; (void)n_in; (void)out_size; (void)ws_size;

  const float* dis_emb = (const float*)d_in[0];
  const float* dr_emb  = (const float*)d_in[1];
  const float* latent  = (const float*)d_in[2];
  const float* di_w    = (const float*)d_in[3];
  const float* dr_w    = (const float*)d_in[4];
  const float* M       = (const float*)d_in[5];   // interact_mat
  const float* Mt      = (const float*)d_in[6];   // interact_mat_t
  const float* U       = (const float*)d_in[7];   // u_edge (propagates dr_sim)
  const float* V       = (const float*)d_in[8];   // v_edge (propagates di_sim)
  const float* di_sim  = (const float*)d_in[9];
  const float* dr_sim  = (const float*)d_in[10];

  float* out      = (float*)d_out;
  float* dis_out  = out;
  float* drug_out = out + (size_t)NN * OUTSTRIDE;
  float* cor_out  = out + (size_t)2 * NN * OUTSTRIDE;

  char* w = (char*)d_ws;
  const size_t EMB = (size_t)NN * DD * sizeof(float);               // 2 MB
  float* wp1_dis = (float*)w; w += EMB;
  float* wp1_dr  = (float*)w; w += EMB;
  unsigned short* XT0 = (unsigned short*)w;
  w += (size_t)4 * NN * DD * sizeof(unsigned short);                // 4 MB
  float* bufA[4]; for (int i = 0; i < 4; ++i) { bufA[i] = (float*)w; w += EMB; }
  float* bufB[4]; for (int i = 0; i < 4; ++i) { bufB[i] = (float*)w; w += EMB; }

  make_wp1<<<2048, 256, 0, stream>>>(latent, di_w, wp1_dis);
  make_wp1<<<2048, 256, 0, stream>>>(latent, dr_w, wp1_dr);
  cor_kernel<<<1, 64, 0, stream>>>(latent, cor_out);

  // hop-0 snapshots
  l2norm_write<<<1024, 256, 0, stream>>>(dis_emb, dis_out + 0);
  l2norm_write<<<1024, 256, 0, stream>>>(di_sim,  dis_out + 64);
  l2norm_write<<<1024, 256, 0, stream>>>(dr_emb,  drug_out + 0);
  l2norm_write<<<1024, 256, 0, stream>>>(dr_sim,  drug_out + 64);

  const float* cur[4] = { dis_emb, dr_emb, di_sim, dr_sim };
  float** nxt = bufA;
  for (int h = 0; h < 3; ++h) {
    // GEMM i: 0 -> M @ dr^T->nxt_dis, 1 -> Mt @ dis^T->nxt_dr,
    //         2 -> V @ di_sim^T,      3 -> U @ dr_sim^T
    Ptr4 srcs;
    srcs.p[0] = cur[1];   // dr_emb      (pre-update)
    srcs.p[1] = cur[0];   // dis_emb     (pre-update)
    srcs.p[2] = cur[2];   // di_sim
    srcs.p[3] = cur[3];   // dr_sim
    transpose4_to_bf16<<<512, 256, 0, stream>>>(srcs, XT0);

    GemmBatch gb;
    gb.A[0] = M;  gb.A[1] = Mt; gb.A[2] = V;  gb.A[3] = U;
    for (int i = 0; i < 4; ++i) gb.XT[i] = XT0 + (size_t)i * NN * DD;
    gb.Y[0] = nxt[0]; gb.Y[1] = nxt[1]; gb.Y[2] = nxt[2]; gb.Y[3] = nxt[3];
    gb.W[0] = wp1_dis; gb.W[1] = wp1_dr; gb.W[2] = nullptr; gb.W[3] = nullptr;
    gemm4_wmma_bf16<<<1024, 256, 0, stream>>>(gb);

    const int base = 128 * (h + 1);
    l2norm_write<<<1024, 256, 0, stream>>>(nxt[0], dis_out  + base);
    l2norm_write<<<1024, 256, 0, stream>>>(nxt[2], dis_out  + base + 64);
    l2norm_write<<<1024, 256, 0, stream>>>(nxt[1], drug_out + base);
    l2norm_write<<<1024, 256, 0, stream>>>(nxt[3], drug_out + base + 64);

    for (int i = 0; i < 4; ++i) cur[i] = nxt[i];
    nxt = (nxt == bufA) ? bufB : bufA;
  }
}
